// RelationProtocolLayer_24180665876975
// MI455X (gfx1250) — matrix operand order repaired and checked by reading, not compiled, not run on hardware
//
#include <hip/hip_runtime.h>

#define NH 12
#define DM 768
#define DH 64
#define BBATCH 2
#define TT 2048
#define MROWS (BBATCH * TT) /* 4096 */
#define SROW 2052           /* padded LDS row stride (floats) to avoid bank conflicts */

typedef __attribute__((ext_vector_type(16))) _Float16 v16h;
typedef __attribute__((ext_vector_type(8)))  float    v8f;

union TileU {
    v16h v;
    _Float16 h[16];
    unsigned int w[8];
};

// ---------------------------------------------------------------------------
// WMMA helpers (CDNA5 / gfx1250, wave32).
// A tile 16x32 f16, row-major source. ISA layout (CDNA5 ISA, 16-bit A 16x32):
//   lane L: row m = L%16, half h = L/16
//   VGPR r: K = (r/4)*16 + h*8 + (r%4)*2, holding K and K+1 packed.
// ---------------------------------------------------------------------------
__device__ inline v16h load_a_f16(const _Float16* A, size_t lda, int m0, int k0, int lane) {
    int m = lane & 15, hf = lane >> 4;
    TileU u;
    const _Float16* row = A + (size_t)(m0 + m) * lda + k0;
#pragma unroll
    for (int r = 0; r < 8; ++r) {
        int kk = ((r >> 2) << 4) + (hf << 3) + ((r & 3) << 1);
        u.w[r] = *(const unsigned int*)(row + kk);
    }
    return u.v;
}

// A tile from fp32 LDS strip rows (rows 0..15 of the strip), converting to f16.
__device__ inline v16h load_a_lds(const float* sc, int k0, int lane) {
    int m = lane & 15, hf = lane >> 4;
    TileU u;
    const float* row = sc + m * SROW + k0;
#pragma unroll
    for (int r = 0; r < 8; ++r) {
        int kk = ((r >> 2) << 4) + (hf << 3) + ((r & 3) << 1);
        u.h[2 * r]     = (_Float16)row[kk];
        u.h[2 * r + 1] = (_Float16)row[kk + 1];
    }
    return u.v;
}

// B tile 32x16 f16, supplied as Bt = B^T row-major (so a B column is a
// contiguous Bt row). ISA B layout (mirroring sparse 64x16 tables):
//   lane L: col n = L%16, half h = L/16
//   VGPR r: K = (r/4)*16 + h*8 + (r%4)*2, holding K and K+1 packed.
__device__ inline v16h load_bT_f16(const _Float16* Bt, size_t ldb, int n0, int k0, int lane) {
    int n = lane & 15, hf = lane >> 4;
    TileU u;
    const _Float16* row = Bt + (size_t)(n0 + n) * ldb + k0;
#pragma unroll
    for (int r = 0; r < 8; ++r) {
        int kk = ((r >> 2) << 4) + (hf << 3) + ((r & 3) << 1);
        u.w[r] = *(const unsigned int*)(row + kk);
    }
    return u.v;
}

__device__ inline v8f wmma_f16(v16h a, v16h b, v8f c) {
    // (neg_a, A, neg_b, B, c_mod, C, reuse_a, reuse_b)
    return __builtin_amdgcn_wmma_f32_16x16x32_f16(false, a, false, b, (short)0, c, false, false);
}

// ---------------------------------------------------------------------------
// Conversion / layout kernels
// ---------------------------------------------------------------------------
__global__ void cvt_f32_to_f16_kernel(const float* __restrict__ in, _Float16* __restrict__ out, int n) {
    int i = blockIdx.x * blockDim.x + threadIdx.x;
    int stride = gridDim.x * blockDim.x;
    for (; i < n; i += stride) out[i] = (_Float16)in[i];
}

// W fp32 [K=768][N=768] row-major -> Wt f16 [N][K]
__global__ void transpose_w_kernel(const float* __restrict__ W, _Float16* __restrict__ Wt) {
    int idx = blockIdx.x * blockDim.x + threadIdx.x; // n*768 + k
    if (idx >= DM * DM) return;
    int n = idx / DM, k = idx % DM;
    Wt[idx] = (_Float16)W[k * DM + n];
}

// rel fp32 [H][d][e] -> relT f16 [H][e][d]
__global__ void transpose_rel_kernel(const float* __restrict__ rel, _Float16* __restrict__ relT) {
    int idx = blockIdx.x * blockDim.x + threadIdx.x; // h*4096 + e*64 + d
    if (idx >= NH * DH * DH) return;
    int h = idx / (DH * DH);
    int rem = idx % (DH * DH);
    int e = rem / DH, d = rem % DH;
    relT[idx] = (_Float16)rel[(h * DH + d) * DH + e];
}

// ---------------------------------------------------------------------------
// Projection GEMM: C[4096x768] = xh[4096x768] @ W  (Wt holds W^T, f16)
// mode 0: scatter to per-head [B,H,T,dh]    (for q, k)
// mode 1: scatter to per-head transposed [B,H,dh,T]  (for v)
// ---------------------------------------------------------------------------
__global__ void proj_kernel(const _Float16* __restrict__ xh, const _Float16* __restrict__ Wt,
                            _Float16* __restrict__ out16, int mode) {
    int lane = threadIdx.x & 31;
    int tile = blockIdx.x * 4 + (threadIdx.x >> 5);
    if (tile >= (MROWS / 16) * (DM / 16)) return;
    int m0 = (tile / (DM / 16)) * 16;
    int n0 = (tile % (DM / 16)) * 16;
    v8f c = {};
    for (int k0 = 0; k0 < DM; k0 += 32) {
        v16h a = load_a_f16(xh, DM, m0, k0, lane);
        v16h b = load_bT_f16(Wt, DM, n0, k0, lane);
        c = wmma_f16(a, b, c);
    }
    int n = lane & 15, hf = lane >> 4;
#pragma unroll
    for (int r = 0; r < 8; ++r) {
        int m = m0 + hf * 8 + r;
        int bb = m >> 11, t = m & (TT - 1);
        int col = n0 + n;
        int h = col >> 6, d = col & 63;
        _Float16 val = (_Float16)c[r];
        if (mode == 0)
            out16[(((size_t)(bb * NH + h) * TT + t) << 6) + d] = val;
        else
            out16[((size_t)(bb * NH + h) * DH + d) * TT + t] = val;
    }
}

// ---------------------------------------------------------------------------
// k_mod[b,h] = k[b,h] (Tx64) @ rel[h] (64x64); Bt = relT[h].
// ---------------------------------------------------------------------------
__global__ void kmod_kernel(const _Float16* __restrict__ k16, const _Float16* __restrict__ relT,
                            _Float16* __restrict__ kmod16) {
    int lane = threadIdx.x & 31;
    int tile = blockIdx.x * 4 + (threadIdx.x >> 5);
    if (tile >= BBATCH * NH * (TT / 16) * (DH / 16)) return;
    int bh = tile / ((TT / 16) * (DH / 16));
    int rem = tile % ((TT / 16) * (DH / 16));
    int m0 = (rem >> 2) * 16;
    int n0 = (rem & 3) * 16;
    int h = bh % NH;
    const _Float16* A  = k16 + (size_t)bh * TT * DH;
    const _Float16* Bt = relT + (size_t)h * DH * DH;
    v8f c = {};
    for (int k0 = 0; k0 < DH; k0 += 32) {
        v16h a = load_a_f16(A, DH, m0, k0, lane);
        v16h b = load_bT_f16(Bt, DH, n0, k0, lane);
        c = wmma_f16(a, b, c);
    }
    _Float16* C = kmod16 + (size_t)bh * TT * DH;
    int n = lane & 15, hf = lane >> 4;
#pragma unroll
    for (int r = 0; r < 8; ++r)
        C[(size_t)(m0 + hf * 8 + r) * DH + n0 + n] = (_Float16)c[r];
}

// ---------------------------------------------------------------------------
// Fused attention strip kernel. One workgroup (8 waves / 256 threads) per
// (b, h, 16-row t-strip). The 16x2048 fp32 score strip lives in LDS (CDNA5:
// 320KB LDS per workgroup). Phases:
//   1) scores = q @ kmod^T / 8 -> LDS strip          (WMMA, per-wave col range)
//   2) row softmax in LDS; coalesced normalized fp32 write to d_out attn
//   3) strip @ v: K split across waves, A from LDS (f32->f16), partials
//      reduced via LDS -> f16 out_heads
// ---------------------------------------------------------------------------
__global__ void __launch_bounds__(256)
attn_strip_kernel(const _Float16* __restrict__ q16, const _Float16* __restrict__ kmod16,
                  const _Float16* __restrict__ vT16, float* __restrict__ attn,
                  _Float16* __restrict__ outh16) {
    __shared__ float sc[16 * SROW];       // 131,328 B score strip
    __shared__ float part[8 * 16 * 64];   //  32,768 B per-wave AV partials
    __shared__ float red[16 * 16];        //   1,024 B row reduction scratch
    __shared__ float rmax[16];
    __shared__ float rinv[16];

    int tid  = threadIdx.x;
    int lane = tid & 31;
    int w    = tid >> 5;

    int bid = blockIdx.x;                 // B*H*(T/16) = 3072
    int bh  = bid / (TT / 16);
    int m0  = (bid % (TT / 16)) * 16;
    int bb  = bh / NH, h = bh % NH;

    // ---------------- Phase 1: scores -> LDS ----------------
    {
        const _Float16* Aq  = q16 + (size_t)bh * TT * DH;
        const _Float16* Btk = kmod16 + (size_t)bh * TT * DH;
        v16h a0 = load_a_f16(Aq, DH, m0, 0, lane);
        v16h a1 = load_a_f16(Aq, DH, m0, 32, lane);
        int n = lane & 15, hf = lane >> 4;
        for (int i = 0; i < 16; ++i) {                 // wave w covers cols [w*256, w*256+256)
            int n0 = (w * 16 + i) * 16;
            v16h b0 = load_bT_f16(Btk, DH, n0, 0, lane);
            v16h b1 = load_bT_f16(Btk, DH, n0, 32, lane);
            v8f c = {};
            c = wmma_f16(a0, b0, c);
            c = wmma_f16(a1, b1, c);
#pragma unroll
            for (int r = 0; r < 8; ++r)
                sc[(hf * 8 + r) * SROW + n0 + n] = c[r] * 0.125f;
        }
    }
    __syncthreads();

    // ---------------- Phase 2: softmax + attn output ----------------
    {
        int r = tid >> 4, c = tid & 15;               // 16 threads per row
        float* row = sc + r * SROW;
        float pm = -1e30f;
        for (int col = c; col < TT; col += 16) pm = fmaxf(pm, row[col]);
        red[r * 16 + c] = pm;
        __syncthreads();
        if (c == 0) {
            float m = red[r * 16];
            for (int i = 1; i < 16; ++i) m = fmaxf(m, red[r * 16 + i]);
            rmax[r] = m;
        }
        __syncthreads();
        float m = rmax[r];
        float ps = 0.0f;
        for (int col = c; col < TT; col += 16) {
            float e = __expf(row[col] - m);
            row[col] = e;
            ps += e;
        }
        red[r * 16 + c] = ps;
        __syncthreads();
        if (c == 0) {
            float s = 0.0f;
            for (int i = 0; i < 16; ++i) s += red[r * 16 + i];
            rinv[r] = 1.0f / s;
        }
        __syncthreads();
        // normalize in LDS + coalesced fp32 write of the attn output strip
        float* gout = attn + ((size_t)bh * TT + m0) * TT;
        for (int idx = tid; idx < 16 * TT; idx += 256) {
            int rr = idx >> 11, col = idx & (TT - 1);
            float v = sc[rr * SROW + col] * rinv[rr];
            sc[rr * SROW + col] = v;
            gout[(size_t)rr * TT + col] = v;
        }
    }
    __syncthreads();

    // ---------------- Phase 3: strip @ v ----------------
    {
        const _Float16* Bt = vT16 + (size_t)bh * DH * TT;
        int n = lane & 15, hf = lane >> 4;
#pragma unroll
        for (int nt = 0; nt < 4; ++nt) {
            int n0 = nt * 16;
            v8f c = {};
            for (int k0 = w * 256; k0 < w * 256 + 256; k0 += 32) {
                v16h a = load_a_lds(sc, k0, lane);
                v16h b = load_bT_f16(Bt, TT, n0, k0, lane);
                c = wmma_f16(a, b, c);
            }
#pragma unroll
            for (int r = 0; r < 8; ++r)
                part[(w * 16 + hf * 8 + r) * 64 + n0 + n] = c[r];
        }
    }
    __syncthreads();
    // reduce 8 wave-partials, emit f16 out_heads [B,T,768]
    for (int idx = tid; idx < 16 * 64; idx += 256) {
        int mrel = idx >> 6, n = idx & 63;
        float s = 0.0f;
#pragma unroll
        for (int ww = 0; ww < 8; ++ww) s += part[(ww * 16 + mrel) * 64 + n];
        outh16[((size_t)(bb * TT) + m0 + mrel) * DM + h * DH + n] = (_Float16)s;
    }
}

// ---------------------------------------------------------------------------
// Final projection: out fp32 [4096x768] = outh16 @ Wo (WoT f16).
// ---------------------------------------------------------------------------
__global__ void outproj_kernel(const _Float16* __restrict__ outh16, const _Float16* __restrict__ WoT,
                               float* __restrict__ out) {
    int lane = threadIdx.x & 31;
    int tile = blockIdx.x * 4 + (threadIdx.x >> 5);
    if (tile >= (MROWS / 16) * (DM / 16)) return;
    int m0 = (tile / (DM / 16)) * 16;
    int n0 = (tile % (DM / 16)) * 16;
    v8f c = {};
    for (int k0 = 0; k0 < DM; k0 += 32) {
        v16h a = load_a_f16(outh16, DM, m0, k0, lane);
        v16h b = load_bT_f16(WoT, DM, n0, k0, lane);
        c = wmma_f16(a, b, c);
    }
    int n = lane & 15, hf = lane >> 4;
#pragma unroll
    for (int r = 0; r < 8; ++r)
        out[(size_t)(m0 + hf * 8 + r) * DM + n0 + n] = c[r];
}

// ---------------------------------------------------------------------------
// Host-side launch
// ---------------------------------------------------------------------------
extern "C" void kernel_launch(void* const* d_in, const int* in_sizes, int n_in,
                              void* d_out, int out_size, void* d_ws, size_t ws_size,
                              hipStream_t stream) {
    const float* x   = (const float*)d_in[0];
    const float* Wq  = (const float*)d_in[1];
    const float* Wk  = (const float*)d_in[2];
    const float* Wv  = (const float*)d_in[3];
    const float* Wo  = (const float*)d_in[4];
    const float* rel = (const float*)d_in[5];

    float* out  = (float*)d_out;                       // [B,T,D]
    float* attn = out + (size_t)MROWS * DM;            // [B,H,T,T]

    // workspace carve-up (f16 buffers, 256B aligned)
    char* base = (char*)d_ws;
    size_t off = 0;
    auto alloc = [&](size_t bytes) {
        void* p = base + off;
        off += (bytes + 255) & ~(size_t)255;
        return p;
    };
    _Float16* xh     = (_Float16*)alloc((size_t)MROWS * DM * 2);
    _Float16* WqT    = (_Float16*)alloc((size_t)DM * DM * 2);
    _Float16* WkT    = (_Float16*)alloc((size_t)DM * DM * 2);
    _Float16* WvT    = (_Float16*)alloc((size_t)DM * DM * 2);
    _Float16* WoT    = (_Float16*)alloc((size_t)DM * DM * 2);
    _Float16* relT   = (_Float16*)alloc((size_t)NH * DH * DH * 2);
    _Float16* q16    = (_Float16*)alloc((size_t)BBATCH * NH * TT * DH * 2);
    _Float16* k16    = (_Float16*)alloc((size_t)BBATCH * NH * TT * DH * 2);
    _Float16* vT16   = (_Float16*)alloc((size_t)BBATCH * NH * DH * TT * 2);
    _Float16* kmod16 = (_Float16*)alloc((size_t)BBATCH * NH * TT * DH * 2);
    _Float16* outh16 = (_Float16*)alloc((size_t)MROWS * DM * 2);
    (void)ws_size;

    // 1) conversions / transposes
    {
        int n = MROWS * DM;
        cvt_f32_to_f16_kernel<<<(n + 255) / 256, 256, 0, stream>>>(x, xh, n);
        int nw = DM * DM;
        transpose_w_kernel<<<(nw + 255) / 256, 256, 0, stream>>>(Wq, WqT);
        transpose_w_kernel<<<(nw + 255) / 256, 256, 0, stream>>>(Wk, WkT);
        transpose_w_kernel<<<(nw + 255) / 256, 256, 0, stream>>>(Wv, WvT);
        transpose_w_kernel<<<(nw + 255) / 256, 256, 0, stream>>>(Wo, WoT);
        int nr = NH * DH * DH;
        transpose_rel_kernel<<<(nr + 255) / 256, 256, 0, stream>>>(rel, relT);
    }

    // 2) q/k/v projections
    {
        int tiles = (MROWS / 16) * (DM / 16);   // 12288
        int blocks = tiles / 4;
        proj_kernel<<<blocks, 128, 0, stream>>>(xh, WqT, q16, 0);
        proj_kernel<<<blocks, 128, 0, stream>>>(xh, WkT, k16, 0);
        proj_kernel<<<blocks, 128, 0, stream>>>(xh, WvT, vT16, 1);
    }

    // 3) per-head bilinear K modulation
    {
        int tiles = BBATCH * NH * (TT / 16) * (DH / 16); // 12288
        kmod_kernel<<<tiles / 4, 128, 0, stream>>>(k16, relT, kmod16);
    }

    // 4) fused scores + softmax + attn-output + (attn @ v)
    attn_strip_kernel<<<BBATCH * NH * (TT / 16), 256, 0, stream>>>(q16, kmod16, vT16, attn, outh16);

    // 5) output projection -> fp32 out
    {
        int tiles = (MROWS / 16) * (DM / 16); // 12288
        outproj_kernel<<<tiles / 4, 128, 0, stream>>>(outh16, WoT, out);
    }
}
